// PriorPromptMaskAttn_4552665334413
// MI455X (gfx1250) — matrix operand order repaired and checked by reading, not compiled, not run on hardware
//
#include <hip/hip_runtime.h>
#include <stdint.h>

#define NQ 100
#define NK 4096
#define DIM 768
#define DH 96
#define KB 64
#define NSEG 4
#define NROW 112   // 7 q-tiles * 16 rows in partial buffers

#define KSTR 104   // Kbuf row stride (bf16): 96+8, keeps 16B align + bank spread
#define VSTR 72    // VbufT row stride: 64+8
#define PSTR 40    // Pbuf row stride: 32+8

typedef __attribute__((ext_vector_type(16))) __bf16 v16bf;
typedef __attribute__((ext_vector_type(8)))  float  v8f;

union Frag16 {
  uint32_t u[8];
  uint4    q[2];
  v16bf    v;
};

__device__ __forceinline__ uint16_t f2bf(float f) {
  uint32_t x = __builtin_bit_cast(uint32_t, f);
  x += 0x7FFFu + ((x >> 16) & 1u);
  return (uint16_t)(x >> 16);
}
__device__ __forceinline__ uint32_t pkbf(float a, float b) {
#if __has_builtin(__builtin_amdgcn_cvt_pk_bf16_f32)
  typedef __attribute__((ext_vector_type(2))) __bf16 v2bf;
  v2bf r = __builtin_amdgcn_cvt_pk_bf16_f32(a, b);
  return __builtin_bit_cast(uint32_t, r);
#else
  return (uint32_t)f2bf(a) | ((uint32_t)f2bf(b) << 16);
#endif
}
__device__ __forceinline__ float fexp2(float x) {
#if __has_builtin(__builtin_amdgcn_exp2f)
  return __builtin_amdgcn_exp2f(x);
#else
  return exp2f(x);
#endif
}
__device__ __forceinline__ float clamp_hi(float x) {
#if __has_builtin(__builtin_amdgcn_fmed3f)
  return __builtin_amdgcn_fmed3f(x, -3.0e38f, 60.0f);   // single v_med3_f32
#else
  return fminf(x, 60.0f);
#endif
}
__device__ __forceinline__ v8f wmma_bf16(v16bf a, v16bf b, v8f c) {
  return __builtin_amdgcn_wmma_f32_16x16x32_bf16(false, a, false, b, (short)0, c,
                                                 false, false);
}

// dst: out[B,NQ,DIM] when lP==nullptr, else accP[NSEG,B,H,NROW,DH] partials
__global__ __launch_bounds__(256, 1) void attn_kernel(
    const float* __restrict__ prompt, const float* __restrict__ prior,
    const int* __restrict__ mask, float* __restrict__ dst,
    float* __restrict__ lP, int klen)
{
  __shared__ __align__(16) uint16_t Kbuf[2][KB * KSTR];    // [key][dim]
  __shared__ __align__(16) uint16_t VbufT[2][DH * VSTR];   // [dim][perm-key]
  __shared__ __align__(16) uint16_t Pbuf[7][16 * PSTR];    // per-wave P staging

  const int seg  = blockIdx.x >> 7;        // 0 for single-pass grid of 128
  const int bh   = blockIdx.x & 127;
  const int b    = bh >> 3;
  const int h    = bh & 7;
  const int tid  = threadIdx.x;
  const int w    = tid >> 5;
  const int lane = tid & 31;
  const int ln   = lane & 15;
  const bool lo  = lane < 16;
  const int kStart = seg * klen;
  const int nblk   = klen / KB;

  // ---- Q as bf16 A-fragments, pre-scaled by 768^-0.5 * log2(e) ----
  const float QSCL = 0.052058783597f;
  Frag16 qA[3];
  {
    int qt = (w < 7) ? w : 6;
    int qr = qt * 16 + ln;
    if (qr > NQ - 1) qr = NQ - 1;
    const float* qp = prompt + (size_t)qr * DIM + h * DH;
    #pragma unroll
    for (int c = 0; c < 3; ++c) {
      int off = c * 32 + (lo ? 0 : 8);
      float4 a0 = *(const float4*)(qp + off);
      float4 a1 = *(const float4*)(qp + off + 4);
      float4 a2 = *(const float4*)(qp + off + 16);
      float4 a3 = *(const float4*)(qp + off + 20);
      qA[c].u[0] = pkbf(a0.x * QSCL, a0.y * QSCL);
      qA[c].u[1] = pkbf(a0.z * QSCL, a0.w * QSCL);
      qA[c].u[2] = pkbf(a1.x * QSCL, a1.y * QSCL);
      qA[c].u[3] = pkbf(a1.z * QSCL, a1.w * QSCL);
      qA[c].u[4] = pkbf(a2.x * QSCL, a2.y * QSCL);
      qA[c].u[5] = pkbf(a2.z * QSCL, a2.w * QSCL);
      qA[c].u[6] = pkbf(a3.x * QSCL, a3.y * QSCL);
      qA[c].u[7] = pkbf(a3.z * QSCL, a3.w * QSCL);
    }
  }

  v8f acc[6];
  #pragma unroll
  for (int t = 0; t < 6; ++t) acc[t] = (v8f)(0.0f);
  float lsum[8];
  #pragma unroll
  for (int r = 0; r < 8; ++r) lsum[r] = 0.0f;

  float4 stage[6];

  auto issue_loads = [&](int k0) {
    const float* src = prior + ((size_t)(b * NK + k0)) * DIM + h * DH;
    #pragma unroll
    for (int i = 0; i < 6; ++i) {
      int f = tid + i * 256;
      int key = f / 24;
      int dg  = f - key * 24;
      stage[i] = *(const float4*)(src + (size_t)key * DIM + dg * 4);
    }
  };
  auto spill_stage = [&](int bbuf) {
    #pragma unroll
    for (int i = 0; i < 6; ++i) {
      int f = tid + i * 256;
      int key = f / 24;
      int dg  = f - key * 24;
      // V columns permuted to match interleaved P packing: j = 2*(k%16) + (k/16)
      int k32  = key & 31;
      int vcol = (key & 32) + (((k32 & 15) << 1) | (k32 >> 4));
      float4 d = stage[i];
      uint2 p;
      p.x = pkbf(d.x, d.y);
      p.y = pkbf(d.z, d.w);
      *(uint2*)&Kbuf[bbuf][key * KSTR + dg * 4] = p;
      VbufT[bbuf][(dg * 4 + 0) * VSTR + vcol] = (uint16_t)(p.x & 0xFFFFu);
      VbufT[bbuf][(dg * 4 + 1) * VSTR + vcol] = (uint16_t)(p.x >> 16);
      VbufT[bbuf][(dg * 4 + 2) * VSTR + vcol] = (uint16_t)(p.y & 0xFFFFu);
      VbufT[bbuf][(dg * 4 + 3) * VSTR + vcol] = (uint16_t)(p.y >> 16);
    }
  };

  issue_loads(kStart);
  spill_stage(0);
  __syncthreads();

  for (int kb = 0; kb < nblk; ++kb) {
    const int bb = kb & 1;
    const int k0 = kStart + kb * KB;
    if (kb + 1 < nblk) issue_loads(k0 + KB);

    if (w < 7) {
      const int* mrow = mask + b * NK + k0;
      int mk[4];
      #pragma unroll
      for (int s = 0; s < 4; ++s) mk[s] = mrow[s * 16 + ln];

      // ---- S = Q*K^T : group all 4 subtile fragments per chunk so one
      //      dscnt wait covers 4 WMMAs instead of draining per WMMA ----
      v8f S[4];
      #pragma unroll
      for (int s = 0; s < 4; ++s) S[s] = (v8f)(0.0f);
      #pragma unroll
      for (int c = 0; c < 3; ++c) {
        Frag16 kf[4];
        #pragma unroll
        for (int s = 0; s < 4; ++s) {
          const uint4* kp =
              (const uint4*)&Kbuf[bb][(s * 16 + ln) * KSTR + c * 32 + (lo ? 0 : 8)];
          kf[s].q[0] = kp[0];
          kf[s].q[1] = kp[2];
        }
        #pragma unroll
        for (int s = 0; s < 4; ++s) S[s] = wmma_bf16(qA[c].v, kf[s].v, S[s]);
      }

      // p = exp2(masked logits); plain row-sum (no max: |logits|~O(2), med3 backstop)
      float p[4][8];
      #pragma unroll
      for (int r = 0; r < 8; ++r) {
        #pragma unroll
        for (int s = 0; s < 4; ++s) {
          float x = mk[s] ? S[s][r] : -1.0e30f;
          p[s][r] = fexp2(clamp_hi(x));
        }
        lsum[r] += (p[0][r] + p[1][r]) + (p[2][r] + p[3][r]);
      }

      uint16_t* pb = Pbuf[w];
      #pragma unroll
      for (int cc = 0; cc < 2; ++cc) {
        // interleaved pair-packed b32 stores (subtile u at lds pos 2n+u)
        #pragma unroll
        for (int r = 0; r < 8; ++r) {
          int row = r + (lo ? 0 : 8);
          *(uint32_t*)&pb[row * PSTR + 2 * ln] =
              pkbf(p[2 * cc][r], p[2 * cc + 1][r]);
        }
        Frag16 pA;
        const uint4* pp = (const uint4*)&pb[ln * PSTR + (lo ? 0 : 16)];
        pA.q[0] = pp[0];
        pA.q[1] = pp[1];
        // group V fragments 3 at a time for the same reason
        #pragma unroll
        for (int tt = 0; tt < 2; ++tt) {
          Frag16 vf[3];
          #pragma unroll
          for (int u = 0; u < 3; ++u) {
            int t = tt * 3 + u;
            const uint4* vp =
                (const uint4*)&VbufT[bb][(t * 16 + ln) * VSTR + cc * 32 + (lo ? 0 : 16)];
            vf[u].q[0] = vp[0];
            vf[u].q[1] = vp[1];
          }
          #pragma unroll
          for (int u = 0; u < 3; ++u)
            acc[tt * 3 + u] = wmma_bf16(pA.v, vf[u].v, acc[tt * 3 + u]);
        }
      }
    }

    __syncthreads();
    if (kb + 1 < nblk) spill_stage(bb ^ 1);
    __syncthreads();
  }

  if (w < 7) {
    float red[8];
    #pragma unroll
    for (int r = 0; r < 8; ++r) {
      float s = lsum[r];
      s += __shfl_xor(s, 1, 32);
      s += __shfl_xor(s, 2, 32);
      s += __shfl_xor(s, 4, 32);
      s += __shfl_xor(s, 8, 32);
      red[r] = s;
    }
    if (lP == nullptr) {
      #pragma unroll
      for (int t = 0; t < 6; ++t) {
        #pragma unroll
        for (int r = 0; r < 8; ++r) {
          int row = w * 16 + r + (lo ? 0 : 8);
          if (row < NQ) {
            dst[((size_t)b * NQ + row) * DIM + h * DH + t * 16 + ln] =
                acc[t][r] / red[r];
          }
        }
      }
    } else {
      const size_t base = (((size_t)seg * 16 + b) * 8 + h) * NROW;
      #pragma unroll
      for (int r = 0; r < 8; ++r) {
        int row = w * 16 + r + (lo ? 0 : 8);
        if (ln == 0) lP[base + row] = red[r];
      }
      #pragma unroll
      for (int t = 0; t < 6; ++t) {
        #pragma unroll
        for (int r = 0; r < 8; ++r) {
          int row = w * 16 + r + (lo ? 0 : 8);
          dst[(base + row) * DH + t * 16 + ln] = acc[t][r];
        }
      }
    }
  }
}

__global__ __launch_bounds__(256) void combine_kernel(
    const float* __restrict__ accP, const float* __restrict__ lP,
    float* __restrict__ out)
{
  int idx = blockIdx.x * 256 + threadIdx.x;           // over 16*100*768
  if (idx >= 16 * NQ * DIM) return;
  int col = idx % DIM;
  int q   = (idx / DIM) % NQ;
  int b   = idx / (DIM * NQ);
  int h   = col / DH;
  int d   = col - h * DH;
  float a = 0.0f, l = 0.0f;
  #pragma unroll
  for (int s = 0; s < NSEG; ++s) {
    size_t base = (((size_t)s * 16 + b) * 8 + h) * NROW + q;
    a += accP[base * DH + d];
    l += lP[base];
  }
  out[idx] = a / l;
}

extern "C" void kernel_launch(void* const* d_in, const int* in_sizes, int n_in,
                              void* d_out, int out_size, void* d_ws, size_t ws_size,
                              hipStream_t stream) {
  const float* prompt = (const float*)d_in[0];  // [100, 768]
  const float* prior  = (const float*)d_in[1];  // [16, 4096, 768]
  const int*   mask   = (const int*)d_in[2];    // [16, 4096]
  float* out = (float*)d_out;                   // [16, 100, 768]

  const size_t accBytes = (size_t)NSEG * 16 * 8 * NROW * DH * sizeof(float);
  const size_t lBytes   = (size_t)NSEG * 16 * 8 * NROW * sizeof(float);

  if (ws_size >= accBytes + lBytes) {
    float* accP = (float*)d_ws;
    float* lP   = (float*)((char*)d_ws + accBytes);
    attn_kernel<<<dim3(NSEG * 128), dim3(256), 0, stream>>>(
        prompt, prior, mask, accP, lP, NK / NSEG);
    combine_kernel<<<dim3((16 * NQ * DIM + 255) / 256), dim3(256), 0, stream>>>(
        accP, lP, out);
  } else {
    attn_kernel<<<dim3(128), dim3(256), 0, stream>>>(
        prompt, prior, mask, out, nullptr, NK);
  }
}
